// GAT_50654844289594
// MI455X (gfx1250) — matrix operand order repaired and checked by reading, not compiled, not run on hardware
//
#include <hip/hip_runtime.h>

// ---------------------------------------------------------------------------
// GAT (3 layers, heads=2, concat=False/head-mean, self-loops) for MI455X.
// Dense xh = h @ W uses v_wmma_f32_16x16x32_f16 (f16 in, f32 accumulate).
// Edge softmax-aggregation fused into 2 atomic passes (max, then exp-weighted
// accumulate with float4/b128 gathers) + per-node normalize/finalize.
//
// Workspace layout (≈117 MB): xh[N,2,128] f32 | acc[N,2,128] f32 |
// h16[N,128] f16 | Wpack[32768] f16 | es,ed,m,denom [N,2] f32.
// ---------------------------------------------------------------------------

#define NNODES   50000
#define NEDGES   800000
#define ETOTAL   (NEDGES + NNODES)   // reference adds self-loops
#define DIM      128
#define HEADS    2
#define OUTW     (HEADS * DIM)       // 256
#define NEG_SLOPE 0.2f

typedef __attribute__((ext_vector_type(16))) _Float16 v16h;
typedef __attribute__((ext_vector_type(8)))  _Float16 v8h;
typedef __attribute__((ext_vector_type(8)))  float    v8f;
typedef __attribute__((ext_vector_type(4)))  float    v4f;

// -------------------------------- helpers ----------------------------------

__device__ __forceinline__ void atomicMaxF(float* addr, float v) {
  // signed-max for v>=0, unsigned-min for v<0; consistent for mixed signs.
  if (v >= 0.0f) atomicMax((int*)addr, __float_as_int(v));
  else           atomicMin((unsigned int*)addr, (unsigned int)__float_as_int(v));
}

__device__ __forceinline__ void edge_sd(const int* __restrict__ ei, int e,
                                        int& s, int& d) {
  if (e < NEDGES) { s = ei[e]; d = ei[NEDGES + e]; }
  else            { s = e - NEDGES; d = s; }        // self-loop
}

__device__ __forceinline__ float leaky(float v) {
  return (v > 0.0f) ? v : NEG_SLOPE * v;
}

// ------------------------------ small kernels ------------------------------

// 8 floats -> 8 halves per thread (b128+b128 load -> b128 store).
__global__ void k_f32_to_f16(const float* __restrict__ src,
                             _Float16* __restrict__ dst, int n8) {
  int i = blockIdx.x * blockDim.x + threadIdx.x;
  if (i >= n8) return;
  v4f lo = *((const v4f*)src + 2 * i);
  v4f hi = *((const v4f*)src + 2 * i + 1);
  v8h o;
#pragma unroll
  for (int j = 0; j < 4; ++j) { o[j] = (_Float16)lo[j]; o[4 + j] = (_Float16)hi[j]; }
  *((v8h*)dst + i) = o;
}

// Pack W[128,256] (row-major f32) into B-fragment lane order (f16).
// Wpack[((ntile*4 + ktile)*32 + lane)*16 + e] =
//   W[(ktile*32 + (lane/16)*8 + (e%8) + (e/8)*16)*256 + ntile*16 + (lane%16)]
__global__ void k_pack_w(const float* __restrict__ W, _Float16* __restrict__ Wp) {
  int i = blockIdx.x * blockDim.x + threadIdx.x;   // 0 .. 32767
  if (i >= 16 * 4 * 32 * 16) return;
  int e    = i & 15;
  int lane = (i >> 4) & 31;
  int kt   = (i >> 9) & 3;
  int nt   = i >> 11;
  int k = kt * 32 + (lane >> 4) * 8 + (e & 7) + (e >> 3) * 16;
  int n = nt * 16 + (lane & 15);
  Wp[i] = (_Float16)W[k * OUTW + n];
}

__global__ void k_init_nodes(float* __restrict__ m, float* __restrict__ den) {
  int i = blockIdx.x * blockDim.x + threadIdx.x;
  if (i < NNODES * HEADS) { m[i] = -INFINITY; den[i] = 0.0f; }
}

// ------------------------------ WMMA GEMM ----------------------------------
// xh[M=50000, 256] = h16[M,128] @ Wpack(128,256).  Block = 4 waves; wave w
// computes a 16x64 tile (4 accumulators).  K loop: 4 steps of 32.

__global__ void __launch_bounds__(128)
k_gemm_wmma(const _Float16* __restrict__ A, const _Float16* __restrict__ Wp,
            float* __restrict__ xh) {
  const int lane    = threadIdx.x & 31;
  const int wave    = threadIdx.x >> 5;     // 0..3 -> 64-col quarter
  const int rowTile = blockIdx.x;           // 0..3124
  const int mrow    = rowTile * 16 + (lane & 15);
  const int kbOff   = (lane < 16) ? 0 : 8;  // 16-bit A fragment lane split
  const int ntBase  = wave * 4;

  v8f c0 = {}, c1 = {}, c2 = {}, c3 = {};
  const v16h* bp = reinterpret_cast<const v16h*>(Wp);

#pragma unroll
  for (int kt = 0; kt < 4; ++kt) {
    const int kb = kt * 32 + kbOff;
    v8h lo = *reinterpret_cast<const v8h*>(A + (size_t)mrow * DIM + kb);
    v8h hi = *reinterpret_cast<const v8h*>(A + (size_t)mrow * DIM + kb + 16);
    v16h a;
#pragma unroll
    for (int i = 0; i < 8; ++i) { a[i] = lo[i]; a[8 + i] = hi[i]; }

    v16h b0 = bp[((ntBase + 0) * 4 + kt) * 32 + lane];
    v16h b1 = bp[((ntBase + 1) * 4 + kt) * 32 + lane];
    v16h b2 = bp[((ntBase + 2) * 4 + kt) * 32 + lane];
    v16h b3 = bp[((ntBase + 3) * 4 + kt) * 32 + lane];

    c0 = __builtin_amdgcn_wmma_f32_16x16x32_f16(false, a, false, b0, (short)0, c0, false, false);
    c1 = __builtin_amdgcn_wmma_f32_16x16x32_f16(false, a, false, b1, (short)0, c1, false, false);
    c2 = __builtin_amdgcn_wmma_f32_16x16x32_f16(false, a, false, b2, (short)0, c2, false, false);
    c3 = __builtin_amdgcn_wmma_f32_16x16x32_f16(false, a, false, b3, (short)0, c3, false, false);
  }

  // C/D layout: VGPR r -> M = r (lanes 0-15) or 8+r (lanes 16-31), N = lane%16
  const int colN = lane & 15;
  const int rOff = (lane < 16) ? 0 : 8;
#pragma unroll
  for (int r = 0; r < 8; ++r) {
    float* o = xh + (size_t)(rowTile * 16 + rOff + r) * OUTW + ntBase * 16;
    o[0 * 16 + colN] = c0[r];
    o[1 * 16 + colN] = c1[r];
    o[2 * 16 + colN] = c2[r];
    o[3 * 16 + colN] = c3[r];
  }
}

// --------------------- attention coefficients es / ed ----------------------
// One wave per (node, head): es[n,h] = dot(xh[n,h,:], a_src[h,:]) etc.

__global__ void __launch_bounds__(256)
k_att_coeff(const float* __restrict__ xh, const float* __restrict__ a_src,
            const float* __restrict__ a_dst, float* __restrict__ es,
            float* __restrict__ ed) {
  int gw   = (blockIdx.x * blockDim.x + threadIdx.x) >> 5;
  int lane = threadIdx.x & 31;
  if (gw >= NNODES * HEADS) return;
  int node = gw >> 1, h = gw & 1;
  const float* xr = xh + (size_t)node * OUTW + h * DIM;
  const float* as = a_src + h * DIM;
  const float* ad = a_dst + h * DIM;
  float s = 0.0f, d = 0.0f;
#pragma unroll
  for (int i = 0; i < 4; ++i) {
    float v = xr[lane + 32 * i];
    s += v * as[lane + 32 * i];
    d += v * ad[lane + 32 * i];
  }
#pragma unroll
  for (int off = 16; off > 0; off >>= 1) {
    s += __shfl_xor(s, off, 32);
    d += __shfl_xor(d, off, 32);
  }
  if (lane == 0) { es[gw] = s; ed[gw] = d; }
}

// ------------------- edge pass 1: per-dst max (both heads) -----------------

__global__ void __launch_bounds__(256)
k_edge_max(const int* __restrict__ ei, const float* __restrict__ es,
           const float* __restrict__ ed, float* __restrict__ m) {
  int e = blockIdx.x * blockDim.x + threadIdx.x;
  if (e >= ETOTAL) return;
  int s, d;
  edge_sd(ei, e, s, d);
  atomicMaxF(&m[d * 2 + 0], leaky(es[s * 2 + 0] + ed[d * 2 + 0]));
  atomicMaxF(&m[d * 2 + 1], leaky(es[s * 2 + 1] + ed[d * 2 + 1]));
}

// ------------- edge pass 2: exp weights -> denom + accumulate --------------
// Softmax normalization is linear: out = (sum_e w_e * xh[src]) / (sum_e w_e).
// 32 threads per edge; each handles 4 channels of both heads via b128 loads.

__global__ void __launch_bounds__(256)
k_edge_agg(const int* __restrict__ ei, const float* __restrict__ es,
           const float* __restrict__ ed, const float* __restrict__ m,
           const float* __restrict__ xh, float* __restrict__ den,
           float* __restrict__ acc) {
  long long gid = (long long)blockIdx.x * blockDim.x + threadIdx.x;
  if (gid >= (long long)ETOTAL * 32) return;
  int e = (int)(gid >> 5);
  int t = (int)(gid & 31);
  int s, d;
  edge_sd(ei, e, s, d);
  const float w0 = __expf(leaky(es[s * 2 + 0] + ed[d * 2 + 0]) - m[d * 2 + 0]);
  const float w1 = __expf(leaky(es[s * 2 + 1] + ed[d * 2 + 1]) - m[d * 2 + 1]);

  v4f x0 = *((const v4f*)(xh + (size_t)s * OUTW) + t);
  v4f x1 = *((const v4f*)(xh + (size_t)s * OUTW + DIM) + t);

  if (t == 0) {
    atomicAdd(&den[d * 2 + 0], w0);
    atomicAdd(&den[d * 2 + 1], w1);
  }
  float* a0 = acc + (size_t)d * OUTW + 4 * t;
  float* a1 = a0 + DIM;
#pragma unroll
  for (int i = 0; i < 4; ++i) {
    atomicAdd(a0 + i, w0 * x0[i]);
    atomicAdd(a1 + i, w1 * x1[i]);
  }
}

// -------------------------- normalize + bias + ELU -------------------------
// 32 threads per node, float4 per thread.

__global__ void __launch_bounds__(256)
k_finalize(const float* __restrict__ acc, const float* __restrict__ den,
           const float* __restrict__ b, float* __restrict__ out, int apply_elu) {
  int i = blockIdx.x * blockDim.x + threadIdx.x;
  if (i >= NNODES * 32) return;
  int n = i >> 5, t = i & 31;
  v4f a0 = *((const v4f*)(acc + (size_t)n * OUTW) + t);
  v4f a1 = *((const v4f*)(acc + (size_t)n * OUTW + DIM) + t);
  v4f bb = *((const v4f*)b + t);
  float r0 = 0.5f / den[n * 2 + 0];
  float r1 = 0.5f / den[n * 2 + 1];
  v4f o;
#pragma unroll
  for (int j = 0; j < 4; ++j) {
    float v = a0[j] * r0 + a1[j] * r1 + bb[j];
    if (apply_elu) v = (v > 0.0f) ? v : (__expf(v) - 1.0f);
    o[j] = v;
  }
  *((v4f*)out + i) = o;
}

// ------------------------------- launcher ----------------------------------

extern "C" void kernel_launch(void* const* d_in, const int* in_sizes, int n_in,
                              void* d_out, int out_size, void* d_ws, size_t ws_size,
                              hipStream_t stream) {
  (void)in_sizes; (void)n_in; (void)out_size; (void)ws_size;

  const float* x  = (const float*)d_in[0];
  const int*   ei = (const int*)d_in[1];
  float*       out = (float*)d_out;

  char* ws = (char*)d_ws;
  size_t off = 0;
  auto carve = [&](size_t bytes) -> char* {
    char* p = ws + off;
    off += (bytes + 255) & ~(size_t)255;
    return p;
  };
  float*    xh  = (float*)carve((size_t)NNODES * OUTW * 4);   // 51.2 MB
  float*    acc = (float*)carve((size_t)NNODES * OUTW * 4);   // 51.2 MB
  _Float16* h16 = (_Float16*)carve((size_t)NNODES * DIM * 2); // 12.8 MB
  _Float16* Wp  = (_Float16*)carve((size_t)32768 * 2);        // 64 KB
  float*    es  = (float*)carve((size_t)NNODES * HEADS * 4);
  float*    ed  = (float*)carve((size_t)NNODES * HEADS * 4);
  float*    m   = (float*)carve((size_t)NNODES * HEADS * 4);
  float*    den = (float*)carve((size_t)NNODES * HEADS * 4);

  const int nFeat8 = NNODES * DIM / 8;   // f32->f16 conversion, 8 elems/thread

  for (int l = 0; l < 3; ++l) {
    const float* W   = (const float*)d_in[2 + 4 * l];
    const float* a_s = (const float*)d_in[3 + 4 * l];
    const float* a_d = (const float*)d_in[4 + 4 * l];
    const float* b   = (const float*)d_in[5 + 4 * l];
    const float* h_in = (l == 0) ? x : out;   // d_out carries features between layers

    k_f32_to_f16<<<(nFeat8 + 255) / 256, 256, 0, stream>>>(h_in, h16, nFeat8);
    k_pack_w<<<(32768 + 255) / 256, 256, 0, stream>>>(W, Wp);
    k_gemm_wmma<<<NNODES / 16, 128, 0, stream>>>(h16, Wp, xh);
    k_att_coeff<<<(NNODES * HEADS * 32 + 255) / 256, 256, 0, stream>>>(xh, a_s, a_d, es, ed);
    k_init_nodes<<<(NNODES * HEADS + 255) / 256, 256, 0, stream>>>(m, den);
    hipMemsetAsync(acc, 0, (size_t)NNODES * OUTW * 4, stream);
    k_edge_max<<<(ETOTAL + 255) / 256, 256, 0, stream>>>(ei, es, ed, m);
    {
      long long tot = (long long)ETOTAL * 32;
      int blocks = (int)((tot + 255) / 256);
      k_edge_agg<<<blocks, 256, 0, stream>>>(ei, es, ed, m, xh, den, acc);
    }
    k_finalize<<<(NNODES * 32 + 255) / 256, 256, 0, stream>>>(acc, den, b, out, (l != 2) ? 1 : 0);
  }
}
